// DeformableDecoder_62371515073112
// MI455X (gfx1250) — compile-verified, matrix operand order checked by reading, not run on hardware
//
#include <hip/hip_runtime.h>

// ---------------------------------------------------------------------------
// Problem constants (from the reference)
// ---------------------------------------------------------------------------
#define D_MODEL 256
#define N_HEAD  8
#define HEAD_D  32
#define N_LVL   4
#define N_PT    4
#define N_LAYER 6
#define D_FF    1024
#define N_QRY   300
#define BATCH   4
#define S_TOT   21760                 // 128^2 + 64^2 + 32^2 + 16^2
#define LQ      (BATCH * N_QRY)       // 1200 query rows
#define MV      (BATCH * S_TOT)       // 87040 value rows

// ---------------------------------------------------------------------------
// WMMA types (CDNA5 / gfx1250, wave32)
// ---------------------------------------------------------------------------
typedef __attribute__((ext_vector_type(16))) __bf16 bf16x16;
typedef __attribute__((ext_vector_type(8)))  float  f32x8;
typedef __attribute__((ext_vector_type(4)))  int    i32x4;

union Frag { i32x4 ld[2]; bf16x16 v; };

__device__ __forceinline__ unsigned short f32_to_bf16_rne(float f) {
  unsigned int u = __float_as_uint(f);
  u += 0x7fffu + ((u >> 16) & 1u);          // round-to-nearest-even
  return (unsigned short)(u >> 16);
}
__device__ __forceinline__ float bf16_to_f32(unsigned short h) {
  return __uint_as_float(((unsigned int)h) << 16);
}

// ---------------------------------------------------------------------------
// f32 -> bf16 conversion (grid-stride)
// ---------------------------------------------------------------------------
__global__ void f32_to_bf16_kernel(const float* __restrict__ in,
                                   unsigned short* __restrict__ out, long n) {
  long i = (long)blockIdx.x * blockDim.x + threadIdx.x;
  long stride = (long)gridDim.x * blockDim.x;
  for (; i < n; i += stride) out[i] = f32_to_bf16_rne(in[i]);
}

// ---------------------------------------------------------------------------
// Generic bf16 WMMA GEMM:  C[M,N] = A[M,K] @ W[N,K]^T + bias[N]
//   ACT: 0 none, 1 relu     OUT_BF16: 0 f32 C, 1 bf16 C (+optional row mask)
// Block = 128 threads = 4 waves, each wave computes a 32x32 tile (2x2 WMMA).
// Requires K % 32 == 0, N % 64 == 0 (true for all uses here); M is clamped.
// ---------------------------------------------------------------------------
template <int ACT, int OUT_BF16>
__global__ __launch_bounds__(128)
void gemm_bf16_kernel(const unsigned short* __restrict__ A,
                      const unsigned short* __restrict__ W,
                      const float* __restrict__ bias,
                      float* __restrict__ Cf,
                      unsigned short* __restrict__ Cb,
                      const unsigned char* __restrict__ rowmask,
                      int M, int N, int K)
{
  const int tid  = threadIdx.x;
  const int wave = tid >> 5;
  const int lane = tid & 31;
  const int half = lane >> 4;     // K-half selector per documented layout
  const int l16  = lane & 15;
  const long rowBase = (long)blockIdx.y * 64 + (wave >> 1) * 32;
  const long colBase = (long)blockIdx.x * 64 + (wave & 1) * 32;
  const int  koff = half * 8;

  f32x8 acc[2][2] = {};

  for (int k0 = 0; k0 < K; k0 += 32) {
    Frag a[2], b[2];
#pragma unroll
    for (int i = 0; i < 2; ++i) {
      long r = rowBase + i * 16 + l16; if (r >= M) r = M - 1;
      const unsigned short* ap = A + r * (long)K + k0 + koff;
      a[i].ld[0] = *reinterpret_cast<const i32x4*>(ap);       // K = k0+koff..+7
      a[i].ld[1] = *reinterpret_cast<const i32x4*>(ap + 16);  // K = +16..+23
      __builtin_prefetch(ap + 32, 0, 0);

      long c = colBase + i * 16 + l16; if (c >= N) c = N - 1;
      const unsigned short* bp = W + c * (long)K + k0 + koff;
      b[i].ld[0] = *reinterpret_cast<const i32x4*>(bp);
      b[i].ld[1] = *reinterpret_cast<const i32x4*>(bp + 16);
      __builtin_prefetch(bp + 32, 0, 0);
    }
#pragma unroll
    for (int i = 0; i < 2; ++i)
#pragma unroll
      for (int j = 0; j < 2; ++j)
        acc[i][j] = __builtin_amdgcn_wmma_f32_16x16x32_bf16(
            false, a[i].v, false, b[j].v, (short)0, acc[i][j], false, false);
  }

#pragma unroll
  for (int i = 0; i < 2; ++i) {
#pragma unroll
    for (int j = 0; j < 2; ++j) {
#pragma unroll
      for (int r = 0; r < 8; ++r) {
        long row = rowBase + i * 16 + r + half * 8;   // C layout: M = vgpr + 8*half
        long col = colBase + j * 16 + l16;
        if (row < M && col < N) {
          float v = acc[i][j][r];
          if (bias) v += bias[col];
          if (ACT == 1) v = v > 0.f ? v : 0.f;
          if (OUT_BF16) {
            if (rowmask && rowmask[row]) v = 0.f;
            Cb[row * (long)N + col] = f32_to_bf16_rne(v);
          } else {
            Cf[row * (long)N + col] = v;
          }
        }
      }
    }
  }
}

// ---------------------------------------------------------------------------
// Reference points: ref = sigmoid(qe @ rp_w^T + rp_b); rpi = ref * valid_ratios
// ---------------------------------------------------------------------------
__global__ void refpoint_kernel(const float* __restrict__ qemb,
                                const float* __restrict__ rp_w,
                                const float* __restrict__ rp_b,
                                const float* __restrict__ valid_ratios,
                                float* __restrict__ ref,
                                float* __restrict__ rpi)
{
  int idx = blockIdx.x * blockDim.x + threadIdx.x;   // b*300 + q
  if (idx >= LQ) return;
  int b = idx / N_QRY, q = idx % N_QRY;
  float sx = rp_b[0], sy = rp_b[1];
  for (int d = 0; d < D_MODEL; ++d) {
    float e = qemb[q * (2 * D_MODEL) + d];
    sx += e * rp_w[d];
    sy += e * rp_w[D_MODEL + d];
  }
  float rx = 1.f / (1.f + __expf(-sx));
  float ry = 1.f / (1.f + __expf(-sy));
  ref[idx * 2 + 0] = rx;
  ref[idx * 2 + 1] = ry;
  for (int lvl = 0; lvl < N_LVL; ++lvl) {
    float vx = valid_ratios[(b * N_LVL + lvl) * 2 + 0];
    float vy = valid_ratios[(b * N_LVL + lvl) * 2 + 1];
    rpi[((long)idx * N_LVL + lvl) * 2 + 0] = rx * vx;
    rpi[((long)idx * N_LVL + lvl) * 2 + 1] = ry * vy;
  }
}

// output init: tgt = query_embed[:, D:]
__global__ void init_output_kernel(const float* __restrict__ qemb,
                                   float* __restrict__ out)
{
  int i = blockIdx.x * blockDim.x + threadIdx.x;
  if (i >= LQ * D_MODEL) return;
  int d = i & (D_MODEL - 1);
  int q = (i >> 8) % N_QRY;
  out[i] = qemb[q * (2 * D_MODEL) + D_MODEL + d];
}

// qin_bf16 = bf16(output + qe)
__global__ void add_qe_bf16_kernel(const float* __restrict__ out,
                                   const float* __restrict__ qemb,
                                   unsigned short* __restrict__ dst)
{
  int i = blockIdx.x * blockDim.x + threadIdx.x;
  if (i >= LQ * D_MODEL) return;
  int d = i & (D_MODEL - 1);
  int q = (i >> 8) % N_QRY;
  dst[i] = f32_to_bf16_rne(out[i] + qemb[q * (2 * D_MODEL) + d]);
}

// ---------------------------------------------------------------------------
// Self-attention core: streaming softmax per (b,h); K,V staged in LDS.
//  qk : (1200, 512)  cols 0..255 = Q (h,d), 256..511 = K (h,d)
//  v  : (1200, 256)
//  o  : (1200, 256)  (b,q,h,d)
// ---------------------------------------------------------------------------
__global__ __launch_bounds__(256)
void attn_kernel(const float* __restrict__ qk,
                 const float* __restrict__ v,
                 float* __restrict__ o)
{
  int b = blockIdx.x >> 3;
  int h = blockIdx.x & 7;
  __shared__ float Ks[N_QRY * HEAD_D];
  __shared__ float Vs[N_QRY * HEAD_D];
  for (int idx = threadIdx.x; idx < N_QRY * HEAD_D; idx += 256) {
    int kq = idx >> 5, d = idx & 31;
    long row = (long)(b * N_QRY + kq);
    Ks[idx] = qk[row * 512 + 256 + h * HEAD_D + d];
    Vs[idx] = v[row * D_MODEL + h * HEAD_D + d];
  }
  __syncthreads();
  const float scale = 0.17677669529663687f;   // 1/sqrt(32)
  for (int q = threadIdx.x; q < N_QRY; q += 256) {
    long qrow = (long)(b * N_QRY + q);
    float qreg[HEAD_D], acc[HEAD_D];
#pragma unroll
    for (int d = 0; d < HEAD_D; ++d) {
      qreg[d] = qk[qrow * 512 + h * HEAD_D + d];
      acc[d] = 0.f;
    }
    float m = -1e30f, l = 0.f;
    for (int kq = 0; kq < N_QRY; ++kq) {
      float s = 0.f;
#pragma unroll
      for (int d = 0; d < HEAD_D; ++d) s += qreg[d] * Ks[kq * HEAD_D + d];
      s *= scale;
      float mn = fmaxf(m, s);
      float corr = __expf(m - mn);
      float p = __expf(s - mn);
      l = l * corr + p;
#pragma unroll
      for (int d = 0; d < HEAD_D; ++d)
        acc[d] = acc[d] * corr + p * Vs[kq * HEAD_D + d];
      m = mn;
    }
    float inv = 1.f / l;
#pragma unroll
    for (int d = 0; d < HEAD_D; ++d)
      o[qrow * D_MODEL + h * HEAD_D + d] = acc[d] * inv;
  }
}

// ---------------------------------------------------------------------------
// out = LayerNorm(x + y) * s + b     (block per row, blockDim == 256 == D)
// ---------------------------------------------------------------------------
__global__ __launch_bounds__(256)
void ln_residual_kernel(const float* __restrict__ x, const float* __restrict__ y,
                        const float* __restrict__ s, const float* __restrict__ b,
                        float* __restrict__ out)
{
  long row = blockIdx.x;
  int d = threadIdx.x;
  float v = x[row * D_MODEL + d] + y[row * D_MODEL + d];
  __shared__ float red[256];
  __shared__ float stats[2];
  red[d] = v; __syncthreads();
  for (int off = 128; off > 0; off >>= 1) {
    if (d < off) red[d] += red[d + off];
    __syncthreads();
  }
  if (d == 0) stats[0] = red[0] * (1.f / D_MODEL);
  __syncthreads();
  float t = v - stats[0];
  red[d] = t * t; __syncthreads();
  for (int off = 128; off > 0; off >>= 1) {
    if (d < off) red[d] += red[d + off];
    __syncthreads();
  }
  if (d == 0) stats[1] = red[0] * (1.f / D_MODEL);
  __syncthreads();
  float inv = rsqrtf(stats[1] + 1e-5f);
  out[row * D_MODEL + d] = t * inv * s[d] + b[d];
}

// softmax over the 16 (lvl,pt) attention-weight logits per (b,q,h)
__global__ void softmax16_kernel(const float* __restrict__ in,
                                 float* __restrict__ out)
{
  int idx = blockIdx.x * blockDim.x + threadIdx.x;   // b*300*8 + q*8 + h
  if (idx >= LQ * N_HEAD) return;
  const float* p = in + (long)(idx >> 3) * 128 + (idx & 7) * 16;
  float*       o = out + (long)(idx >> 3) * 128 + (idx & 7) * 16;
  float m = -1e30f;
#pragma unroll
  for (int j = 0; j < 16; ++j) m = fmaxf(m, p[j]);
  float e[16], sum = 0.f;
#pragma unroll
  for (int j = 0; j < 16; ++j) { e[j] = __expf(p[j] - m); sum += e[j]; }
  float inv = 1.f / sum;
#pragma unroll
  for (int j = 0; j < 16; ++j) o[j] = e[j] * inv;
}

// ---------------------------------------------------------------------------
// Deformable bilinear sampling. Block per (b,q); thread = (head, dim).
//  vbf  : (bs, S, 256) projected values, bf16
//  off  : (1200, 256)  sampling offsets  (h, lvl, pt, 2)
//  aw   : (1200, 128)  softmaxed weights (h, lvl*pt)
//  rpi  : (1200, 4, 2)
// ---------------------------------------------------------------------------
__global__ __launch_bounds__(256)
void deform_sample_kernel(const unsigned short* __restrict__ vbf,
                          const float* __restrict__ off,
                          const float* __restrict__ aw,
                          const float* __restrict__ rpi,
                          float* __restrict__ outp)
{
  int bq = blockIdx.x;
  int b  = bq / N_QRY;
  int h  = threadIdx.x >> 5;
  int d  = threadIdx.x & 31;
  const int HWs[4][2] = {{128, 128}, {64, 64}, {32, 32}, {16, 16}};
  const int startv[4] = {0, 16384, 20480, 21504};
  const float* offp = off + (long)bq * D_MODEL + h * (N_LVL * N_PT * 2);
  const float* awp  = aw  + (long)bq * (N_HEAD * N_LVL * N_PT) + h * (N_LVL * N_PT);
  float acc = 0.f;
#pragma unroll
  for (int lvl = 0; lvl < N_LVL; ++lvl) {
    const int H = HWs[lvl][0], W = HWs[lvl][1];
    float rx = rpi[((long)bq * N_LVL + lvl) * 2 + 0];
    float ry = rpi[((long)bq * N_LVL + lvl) * 2 + 1];
    long base = ((long)b * S_TOT + startv[lvl]) * (long)D_MODEL + h * HEAD_D + d;
#pragma unroll
    for (int p = 0; p < N_PT; ++p) {
      float gx = rx * W + offp[lvl * (N_PT * 2) + p * 2 + 0] - 0.5f;
      float gy = ry * H + offp[lvl * (N_PT * 2) + p * 2 + 1] - 0.5f;
      float a  = awp[lvl * N_PT + p];
      float x0 = floorf(gx), y0 = floorf(gy);
      float samp = 0.f;
#pragma unroll
      for (int dx = 0; dx <= 1; ++dx)
#pragma unroll
        for (int dy = 0; dy <= 1; ++dy) {
          float xi = x0 + dx, yi = y0 + dy;
          float wgt = (1.f - fabsf(gx - xi)) * (1.f - fabsf(gy - yi));
          bool valid = (xi >= 0.f) && (xi < (float)W) && (yi >= 0.f) && (yi < (float)H);
          float xc = fminf(fmaxf(xi, 0.f), (float)(W - 1));
          float yc = fminf(fmaxf(yi, 0.f), (float)(H - 1));
          int idx = (int)yc * W + (int)xc;
          float g = bf16_to_f32(vbf[base + (long)idx * D_MODEL]);
          samp += valid ? wgt * g : 0.f;
        }
      acc += a * samp;
    }
  }
  outp[(long)bq * D_MODEL + h * HEAD_D + d] = acc;
}

// ---------------------------------------------------------------------------
// Host orchestration
// ---------------------------------------------------------------------------
static void gemm(const unsigned short* A, const unsigned short* W, const float* bias,
                 void* C, const unsigned char* rowmask,
                 int M, int N, int K, int act, int outbf, hipStream_t s)
{
  dim3 grid((N + 63) / 64, (M + 63) / 64), block(128);
  if (!outbf)
    gemm_bf16_kernel<0, 0><<<grid, block, 0, s>>>(A, W, bias, (float*)C, nullptr, nullptr, M, N, K);
  else if (act == 0)
    gemm_bf16_kernel<0, 1><<<grid, block, 0, s>>>(A, W, bias, nullptr, (unsigned short*)C, rowmask, M, N, K);
  else
    gemm_bf16_kernel<1, 1><<<grid, block, 0, s>>>(A, W, bias, nullptr, (unsigned short*)C, rowmask, M, N, K);
}

extern "C" void kernel_launch(void* const* d_in, const int* in_sizes, int n_in,
                              void* d_out, int out_size, void* d_ws, size_t ws_size,
                              hipStream_t stream)
{
  (void)in_sizes; (void)n_in; (void)out_size; (void)ws_size;

  const float* memory        = (const float*)d_in[0];
  const unsigned char* mask  = (const unsigned char*)d_in[1];
  const float* valid_ratios  = (const float*)d_in[3];
  const float* query_embed   = (const float*)d_in[4];
  const float* rp_w  = (const float*)d_in[5];
  const float* rp_b  = (const float*)d_in[6];
  const float* so_w  = (const float*)d_in[7];
  const float* so_b  = (const float*)d_in[8];
  const float* aw_w  = (const float*)d_in[9];
  const float* aw_b  = (const float*)d_in[10];
  const float* vp_w  = (const float*)d_in[11];
  const float* vp_b  = (const float*)d_in[12];
  const float* op_w  = (const float*)d_in[13];
  const float* op_b  = (const float*)d_in[14];
  const float* sa_in_w  = (const float*)d_in[15];
  const float* sa_in_b  = (const float*)d_in[16];
  const float* sa_out_w = (const float*)d_in[17];
  const float* sa_out_b = (const float*)d_in[18];
  const float* n1_s = (const float*)d_in[19];
  const float* n1_b = (const float*)d_in[20];
  const float* n2_s = (const float*)d_in[21];
  const float* n2_b = (const float*)d_in[22];
  const float* n3_s = (const float*)d_in[23];
  const float* n3_b = (const float*)d_in[24];
  const float* l1_w = (const float*)d_in[25];
  const float* l1_b = (const float*)d_in[26];
  const float* l2_w = (const float*)d_in[27];
  const float* l2_b = (const float*)d_in[28];

  char* base = (char*)d_ws;
  size_t cur = 0;
  auto alloc = [&](size_t bytes) -> void* {
    cur = (cur + 255) & ~(size_t)255;
    void* p = base + cur;
    cur += bytes;
    return p;
  };

  // bf16 staging buffers
  unsigned short* mem_bf   = (unsigned short*)alloc((size_t)MV * D_MODEL * 2);
  unsigned short* v_bf     = (unsigned short*)alloc((size_t)MV * D_MODEL * 2);
  unsigned short* w_sa_in  = (unsigned short*)alloc((size_t)N_LAYER * 3 * D_MODEL * D_MODEL * 2);
  unsigned short* w_sa_out = (unsigned short*)alloc((size_t)N_LAYER * D_MODEL * D_MODEL * 2);
  unsigned short* w_so     = (unsigned short*)alloc((size_t)N_LAYER * D_MODEL * D_MODEL * 2);
  unsigned short* w_aw     = (unsigned short*)alloc((size_t)N_LAYER * 128 * D_MODEL * 2);
  unsigned short* w_vp     = (unsigned short*)alloc((size_t)N_LAYER * D_MODEL * D_MODEL * 2);
  unsigned short* w_op     = (unsigned short*)alloc((size_t)N_LAYER * D_MODEL * D_MODEL * 2);
  unsigned short* w_l1     = (unsigned short*)alloc((size_t)N_LAYER * D_FF * D_MODEL * 2);
  unsigned short* w_l2     = (unsigned short*)alloc((size_t)N_LAYER * D_MODEL * D_FF * 2);
  // f32 buffers
  float* output  = (float*)alloc((size_t)LQ * D_MODEL * 4);
  float* tmp256  = (float*)alloc((size_t)LQ * D_MODEL * 4);
  float* qk      = (float*)alloc((size_t)LQ * 512 * 4);
  float* vsa     = (float*)alloc((size_t)LQ * D_MODEL * 4);
  float* atto    = (float*)alloc((size_t)LQ * D_MODEL * 4);
  float* so_out  = (float*)alloc((size_t)LQ * D_MODEL * 4);
  float* aw_out  = (float*)alloc((size_t)LQ * 128 * 4);
  float* aw_soft = (float*)alloc((size_t)LQ * 128 * 4);
  float* samp    = (float*)alloc((size_t)LQ * D_MODEL * 4);
  float* refbuf  = (float*)alloc((size_t)LQ * 2 * 4);
  float* rpi     = (float*)alloc((size_t)LQ * N_LVL * 2 * 4);
  // bf16 activation buffers
  unsigned short* qin_bf  = (unsigned short*)alloc((size_t)LQ * D_MODEL * 2);
  unsigned short* out_bf  = (unsigned short*)alloc((size_t)LQ * D_MODEL * 2);
  unsigned short* atto_bf = (unsigned short*)alloc((size_t)LQ * D_MODEL * 2);
  unsigned short* samp_bf = (unsigned short*)alloc((size_t)LQ * D_MODEL * 2);
  unsigned short* l1_bf   = (unsigned short*)alloc((size_t)LQ * D_FF * 2);

  auto convert = [&](const float* in, unsigned short* out, long n) {
    long blocks = (n + 255) / 256;
    if (blocks > 4096) blocks = 4096;
    f32_to_bf16_kernel<<<(int)blocks, 256, 0, stream>>>(in, out, n);
  };

  // ---- one-time staging ----
  convert(memory, mem_bf, (long)MV * D_MODEL);
  convert(sa_in_w,  w_sa_in,  (long)N_LAYER * 3 * D_MODEL * D_MODEL);
  convert(sa_out_w, w_sa_out, (long)N_LAYER * D_MODEL * D_MODEL);
  convert(so_w,     w_so,     (long)N_LAYER * D_MODEL * D_MODEL);
  convert(aw_w,     w_aw,     (long)N_LAYER * 128 * D_MODEL);
  convert(vp_w,     w_vp,     (long)N_LAYER * D_MODEL * D_MODEL);
  convert(op_w,     w_op,     (long)N_LAYER * D_MODEL * D_MODEL);
  convert(l1_w,     w_l1,     (long)N_LAYER * D_FF * D_MODEL);
  convert(l2_w,     w_l2,     (long)N_LAYER * D_MODEL * D_FF);

  refpoint_kernel<<<(LQ + 127) / 128, 128, 0, stream>>>(query_embed, rp_w, rp_b,
                                                        valid_ratios, refbuf, rpi);
  init_output_kernel<<<(LQ * D_MODEL + 255) / 256, 256, 0, stream>>>(query_embed, output);

  const int NE = LQ * D_MODEL;

  for (int lid = 0; lid < N_LAYER; ++lid) {
    // ---------------- self-attention ----------------
    add_qe_bf16_kernel<<<(NE + 255) / 256, 256, 0, stream>>>(output, query_embed, qin_bf);
    convert(output, out_bf, NE);
    // Q,K projection (rows 0..511 of sa_in_w)
    gemm(qin_bf, w_sa_in + (size_t)lid * 3 * D_MODEL * D_MODEL,
         sa_in_b + (size_t)lid * 3 * D_MODEL,
         qk, nullptr, LQ, 512, D_MODEL, 0, 0, stream);
    // V projection (rows 512..767)
    gemm(out_bf, w_sa_in + (size_t)lid * 3 * D_MODEL * D_MODEL + (size_t)512 * D_MODEL,
         sa_in_b + (size_t)lid * 3 * D_MODEL + 512,
         vsa, nullptr, LQ, D_MODEL, D_MODEL, 0, 0, stream);
    attn_kernel<<<BATCH * N_HEAD, 256, 0, stream>>>(qk, vsa, atto);
    convert(atto, atto_bf, NE);
    gemm(atto_bf, w_sa_out + (size_t)lid * D_MODEL * D_MODEL,
         sa_out_b + (size_t)lid * D_MODEL,
         tmp256, nullptr, LQ, D_MODEL, D_MODEL, 0, 0, stream);
    ln_residual_kernel<<<LQ, 256, 0, stream>>>(output, tmp256,
        n2_s + (size_t)lid * D_MODEL, n2_b + (size_t)lid * D_MODEL, output);

    // ---------------- deformable cross-attention ----------------
    add_qe_bf16_kernel<<<(NE + 255) / 256, 256, 0, stream>>>(output, query_embed, qin_bf);
    gemm(qin_bf, w_so + (size_t)lid * D_MODEL * D_MODEL,
         so_b + (size_t)lid * D_MODEL,
         so_out, nullptr, LQ, D_MODEL, D_MODEL, 0, 0, stream);
    gemm(qin_bf, w_aw + (size_t)lid * 128 * D_MODEL,
         aw_b + (size_t)lid * 128,
         aw_out, nullptr, LQ, 128, D_MODEL, 0, 0, stream);
    softmax16_kernel<<<(LQ * N_HEAD + 255) / 256, 256, 0, stream>>>(aw_out, aw_soft);
    // value projection over all of `memory` -> bf16 (mask rows zeroed)
    gemm(mem_bf, w_vp + (size_t)lid * D_MODEL * D_MODEL,
         vp_b + (size_t)lid * D_MODEL,
         v_bf, mask, MV, D_MODEL, D_MODEL, 0, 1, stream);
    deform_sample_kernel<<<LQ, 256, 0, stream>>>(v_bf, so_out, aw_soft, rpi, samp);
    convert(samp, samp_bf, NE);
    gemm(samp_bf, w_op + (size_t)lid * D_MODEL * D_MODEL,
         op_b + (size_t)lid * D_MODEL,
         tmp256, nullptr, LQ, D_MODEL, D_MODEL, 0, 0, stream);
    ln_residual_kernel<<<LQ, 256, 0, stream>>>(output, tmp256,
        n1_s + (size_t)lid * D_MODEL, n1_b + (size_t)lid * D_MODEL, output);

    // ---------------- FFN ----------------
    convert(output, out_bf, NE);
    gemm(out_bf, w_l1 + (size_t)lid * D_FF * D_MODEL,
         l1_b + (size_t)lid * D_FF,
         l1_bf, nullptr, LQ, D_FF, D_MODEL, /*relu*/1, /*bf16*/1, stream);
    gemm(l1_bf, w_l2 + (size_t)lid * D_MODEL * D_FF,
         l2_b + (size_t)lid * D_MODEL,
         tmp256, nullptr, LQ, D_MODEL, D_FF, 0, 0, stream);
    float* dst = (lid == N_LAYER - 1) ? (float*)d_out : output;
    ln_residual_kernel<<<LQ, 256, 0, stream>>>(output, tmp256,
        n3_s + (size_t)lid * D_MODEL, n3_b + (size_t)lid * D_MODEL, dst);
  }
}
